// Transformer_20641612825307
// MI455X (gfx1250) — matrix operand order repaired and checked by reading, not compiled, not run on hardware
//
#include <hip/hip_runtime.h>
#include <hip/hip_bf16.h>
#include <math.h>

// ---------------------------------------------------------------------------
// Types
// ---------------------------------------------------------------------------
typedef __attribute__((ext_vector_type(16))) __bf16 v16bf;
typedef __attribute__((ext_vector_type(8)))  __bf16 v8bf;
typedef __attribute__((ext_vector_type(8)))  float  v8f;

#define BM 128
#define BN 128
#define BKK 32
#define LDP 40   // padded LDS row (bf16 elems): 80B stride, conflict-free, 16B aligned

#if __has_builtin(__builtin_amdgcn_tensor_load_to_lds) && \
    __has_builtin(__builtin_amdgcn_s_wait_tensorcnt)
#define USE_TDM 1
#else
#define USE_TDM 0
#endif

// Two aligned 16B reads -> one 16-elem bf16 fragment (works for LDS or global).
__device__ __forceinline__ v16bf frag_ld(const __bf16* row, int e0, int e1) {
  union { v16bf v; v8bf h[2]; } u;
  u.h[0] = *(const v8bf*)(row + e0);
  u.h[1] = *(const v8bf*)(row + e1);
  return u.v;
}

#if USE_TDM
typedef __attribute__((ext_vector_type(4))) unsigned int u32x4;
typedef __attribute__((ext_vector_type(8))) int          i32x8;
typedef __attribute__((ext_vector_type(4))) int          i32x4;

// Byte offset of a __shared__ object inside LDS (generic -> local addrspace).
__device__ __forceinline__ unsigned lds_offset(const void* p) {
  return (unsigned)(unsigned long long)
      (const __attribute__((address_space(3))) char*)p;
}

// Issue one TDM 2-D tile load: tile_rows x tile_ex bf16 elements, global row
// stride = stride_e elems, into LDS with a 16B pad every 64B (-> [rows][40]).
// Descriptor bit layout per CDNA5 ISA 8.3 (group 0) / 8.4 (group 1).
// This toolchain's builtin takes 6 args: (g0, g1, g2, g3, g_extra, cpol).
__device__ __forceinline__ void tdm_load_2d(
    const __bf16* gsrc, unsigned lds_byte_addr,
    unsigned tile_ex, unsigned tile_rows, unsigned stride_e)
{
  unsigned long long ga = (unsigned long long)gsrc;
  u32x4 g0;
  g0[0] = 1u;                                            // count=1 (valid), user
  g0[1] = lds_byte_addr;                                 // lds_addr
  g0[2] = (unsigned)(ga & 0xFFFFFFFFu);                  // global_addr lo
  g0[3] = (unsigned)((ga >> 32) & 0x01FFFFFFu)           // global_addr hi (57b)
        | (2u << 30);                                    // type = 2 ("image")
  i32x8 g1;
  g1[0] = (int)((1u << 16)      // data_size = 1 -> 2 bytes
              | (1u << 20)      // pad_enable
              | (3u << 22)      // pad_interval = 3 -> every 16 DWORDs (64B)
              | (3u << 25));    // pad_amount   = 3 -> 4 DWORDs (16B)
  g1[1] = (int)((tile_ex & 0xFFFFu) << 16);              // tensor_dim0[15:0]
  g1[2] = (int)((tile_ex >> 16) | ((tile_rows & 0xFFFFu) << 16)); // td0 hi | td1 lo
  g1[3] = (int)((tile_rows >> 16) | ((tile_ex & 0xFFFFu) << 16)); // td1 hi | tile_dim0
  g1[4] = (int)(tile_rows & 0xFFFFu);                    // tile_dim1 (tile_dim2=0)
  g1[5] = (int)stride_e;                                 // tensor_dim0_stride lo
  g1[6] = 0;                                             // stride hi | dim1_stride lo
  g1[7] = 0;
  i32x4 gz;  gz[0] = gz[1] = gz[2] = gz[3] = 0;
  i32x8 gz8;
  #pragma unroll
  for (int i = 0; i < 8; ++i) gz8[i] = 0;
  __builtin_amdgcn_tensor_load_to_lds(g0, g1, gz, gz, gz8, 0);
}
#endif

// ---------------------------------------------------------------------------
// bf16 WMMA GEMM: C = A[M,K](bf16) @ W[K,ldw](bf16) (+bias)(+res fp32)(relu?)
// A tile staged by the Tensor Data Mover (wave 0 issues, TENSORcnt wait);
// W tile transposed into LDS by all 8 waves. Wave tile 32x64.
// ---------------------------------------------------------------------------
template<bool RELU>
__global__ __launch_bounds__(256) void gemm_bf16_wmma(
    const __bf16* __restrict__ A, const __bf16* __restrict__ W,
    const float* __restrict__ bias, const float* __restrict__ res,
    float* __restrict__ Cf, __bf16* __restrict__ Cb,
    int M, int N, int K, int ldw)
{
  __shared__ __bf16 As[BM][LDP];    // (m, k)
  __shared__ __bf16 Bst[BN][LDP];   // (n, k)  = W tile transposed

  const int tid  = threadIdx.x;
  const int bm   = blockIdx.y * BM;
  const int bn   = blockIdx.x * BN;
  const int wid  = tid >> 5;
  const int lane = tid & 31;
  const int half = lane >> 4;
  const int l15  = lane & 15;
  const int wm   = (wid >> 1) * 32;
  const int wn   = (wid & 1) * 64;

#if USE_TDM
  const unsigned as_base = lds_offset(&As[0][0]);
#endif

  v8f acc[2][4];
  #pragma unroll
  for (int i = 0; i < 2; ++i)
    #pragma unroll
    for (int j = 0; j < 4; ++j)
      #pragma unroll
      for (int e = 0; e < 8; ++e) acc[i][j][e] = 0.0f;

  for (int k0 = 0; k0 < K; k0 += BKK) {
#if USE_TDM
    // ---- A tile via Tensor Data Mover: one issue per block ----
    if (wid == 0)
      tdm_load_2d(A + (size_t)bm * K + k0, as_base, BKK, BM, (unsigned)K);
#else
    // ---- fallback: synchronous A staging, 16B vector copies ----
    #pragma unroll
    for (int it = 0; it < 2; ++it) {
      int lin = tid + it * 256;
      int r   = lin >> 2;
      int c8  = (lin & 3) * 8;
      int gr  = bm + r;
      v8bf av;
      if (gr < M) {
        av = *(const v8bf*)(A + (size_t)gr * K + k0 + c8);
      } else {
        #pragma unroll
        for (int j = 0; j < 8; ++j) av[j] = (__bf16)0.0f;
      }
      *(v8bf*)(&As[r][c8]) = av;
    }
#endif
    // ---- stage W tile (32x128 bf16) transposed; ldw padded to mult of 8 ----
    #pragma unroll
    for (int it = 0; it < 2; ++it) {
      int lin = tid + it * 256;
      int r   = lin >> 4;                // k row 0..31
      int c8  = (lin & 15) * 8;          // 0..120
      const __bf16* wp = W + (size_t)(k0 + r) * ldw + bn + c8;
      v8bf wv;
      if (bn + c8 + 8 <= ldw) {
        wv = *(const v8bf*)wp;
      } else {
        #pragma unroll
        for (int j = 0; j < 8; ++j)
          wv[j] = (bn + c8 + j < ldw) ? wp[j] : (__bf16)0.0f;
      }
      #pragma unroll
      for (int j = 0; j < 8; ++j) Bst[c8 + j][r] = wv[j];
    }
    // prefetch next W tile (global_prefetch_b8)
    if (k0 + BKK < K) {
      int r = tid >> 3;
      if (r < BKK) {
        int cv = (tid & 7) * 16;
        if (bn + cv < ldw)
          __builtin_prefetch(W + (size_t)(k0 + BKK + r) * ldw + bn + cv, 0, 1);
      }
    }
#if USE_TDM
    if (wid == 0) __builtin_amdgcn_s_wait_tensorcnt(0);
#endif
    __syncthreads();

    // ---- fragments per ISA 7.12.2 layouts ----
    v16bf af[2], bfr[4];
    #pragma unroll
    for (int mi = 0; mi < 2; ++mi)
      af[mi] = frag_ld(&As[wm + mi * 16 + l15][0], half * 8, half * 8 + 16);
    #pragma unroll
    for (int ni = 0; ni < 4; ++ni)
      bfr[ni] = frag_ld(&Bst[wn + ni * 16 + l15][0], half * 8, half * 8 + 16);

    #pragma unroll
    for (int mi = 0; mi < 2; ++mi)
      #pragma unroll
      for (int ni = 0; ni < 4; ++ni)
        acc[mi][ni] = __builtin_amdgcn_wmma_f32_16x16x32_bf16(
            false, af[mi], false, bfr[ni], (short)0, acc[mi][ni], false, false);

    __syncthreads();
  }

  // ---- epilogue ----
  #pragma unroll
  for (int mi = 0; mi < 2; ++mi) {
    #pragma unroll
    for (int ni = 0; ni < 4; ++ni) {
      int col = bn + wn + ni * 16 + l15;
      if (col >= N) continue;
      #pragma unroll
      for (int r = 0; r < 8; ++r) {
        int row = bm + wm + mi * 16 + half * 8 + r;
        if (row >= M) continue;
        float v = acc[mi][ni][r];
        if (bias) v += bias[col];
        if (res)  v += res[(size_t)row * N + col];
        if (RELU) v = fmaxf(v, 0.0f);
        if (Cf) Cf[(size_t)row * N + col] = v;
        if (Cb) Cb[(size_t)row * N + col] = (__bf16)v;
      }
    }
  }
}

// ---------------------------------------------------------------------------
// Fused flash attention (WMMA). Block = 8 waves; wave owns 16 q rows; 32-key
// tiles with online softmax. Q@K^T fragments load STRAIGHT from global bf16;
// only V^T is LDS-staged.
// ---------------------------------------------------------------------------
__global__ __launch_bounds__(256) void flash_attn_wmma(
    const __bf16* __restrict__ Qb, const __bf16* __restrict__ Kb,
    const __bf16* __restrict__ Vb, const float* __restrict__ mask,
    __bf16* __restrict__ Ob, int L, int E, int DK, float scale)
{
  __shared__ __bf16 Vt[64][LDP];            // [d][key] transposed V tile
  __shared__ float  Sw[8][16][36];          // per-wave score tile (fp32)
  __shared__ __bf16 Pw[8][16][LDP];         // per-wave prob tile (bf16)
  __shared__ float  mrow[8][16], crow[8][16], lrow[8][16];

  const int tid  = threadIdx.x;
  const int wid  = tid >> 5;
  const int lane = tid & 31;
  const int half = lane >> 4;
  const int l15  = lane & 15;
  const int h = blockIdx.y, b = blockIdx.z;
  const int q0 = blockIdx.x * 128 + wid * 16;

  v16bf qf[2];
  {
    const __bf16* qrow = Qb + ((size_t)b * L + q0 + l15) * E + h * DK;
    #pragma unroll
    for (int ks = 0; ks < 2; ++ks) {
      int be = ks * 32 + half * 8;
      qf[ks] = frag_ld(qrow, be, be + 16);
    }
  }

  v8f oacc[4];
  #pragma unroll
  for (int ni = 0; ni < 4; ++ni)
    #pragma unroll
    for (int e = 0; e < 8; ++e) oacc[ni][e] = 0.0f;
  if (lane < 16) { mrow[wid][l15] = -3.4e38f; lrow[wid][l15] = 0.0f; }

  for (int kt = 0; kt < L; kt += 32) {
    __syncthreads();
    {
      int key = tid >> 3;
      int d0  = (tid & 7) * 8;
      v8bf vv = *(const v8bf*)(Vb + ((size_t)b * L + kt + key) * E + h * DK + d0);
      #pragma unroll
      for (int j = 0; j < 8; ++j) Vt[d0 + j][key] = vv[j];
    }
    v8f sacc[2];
    #pragma unroll
    for (int ni = 0; ni < 2; ++ni)
      #pragma unroll
      for (int e = 0; e < 8; ++e) sacc[ni][e] = 0.0f;
    #pragma unroll
    for (int ni = 0; ni < 2; ++ni) {
      const __bf16* krow = Kb + ((size_t)b * L + kt + ni * 16 + l15) * E + h * DK;
      #pragma unroll
      for (int ks = 0; ks < 2; ++ks) {
        int be = ks * 32 + half * 8;
        v16bf kf = frag_ld(krow, be, be + 16);
        sacc[ni] = __builtin_amdgcn_wmma_f32_16x16x32_bf16(
            false, qf[ks], false, kf, (short)0, sacc[ni], false, false);
      }
    }
    #pragma unroll
    for (int ni = 0; ni < 2; ++ni) {
      int col = ni * 16 + l15;
      #pragma unroll
      for (int r = 0; r < 8; ++r) {
        int row = half * 8 + r;
        float s = sacc[ni][r] * scale;
        float m = mask[((size_t)b * L + q0 + row) * L + kt + col];
        Sw[wid][row][col] = (m == 0.0f) ? -1e9f : s;
      }
    }
    __syncthreads();
    if (lane < 16) {
      int r = l15;
      float mold = mrow[wid][r];
      float tm = mold;
      for (int c = 0; c < 32; ++c) tm = fmaxf(tm, Sw[wid][r][c]);
      float corr = __expf(mold - tm);
      float ssum = 0.0f;
      for (int c = 0; c < 32; ++c) {
        float p = __expf(Sw[wid][r][c] - tm);
        Pw[wid][r][c] = (__bf16)p;
        ssum += p;
      }
      mrow[wid][r] = tm;
      crow[wid][r] = corr;
      lrow[wid][r] = lrow[wid][r] * corr + ssum;
    }
    __syncthreads();
    v16bf pf = frag_ld(&Pw[wid][l15][0], half * 8, half * 8 + 16);
    #pragma unroll
    for (int ni = 0; ni < 4; ++ni) {
      #pragma unroll
      for (int r = 0; r < 8; ++r) oacc[ni][r] *= crow[wid][half * 8 + r];
      v16bf vf = frag_ld(&Vt[ni * 16 + l15][0], half * 8, half * 8 + 16);
      oacc[ni] = __builtin_amdgcn_wmma_f32_16x16x32_bf16(
          false, pf, false, vf, (short)0, oacc[ni], false, false);
    }
  }

  #pragma unroll
  for (int ni = 0; ni < 4; ++ni) {
    int d = ni * 16 + l15;
    #pragma unroll
    for (int r = 0; r < 8; ++r) {
      int row = half * 8 + r;
      Ob[((size_t)b * L + q0 + row) * E + h * DK + d] =
          (__bf16)(oacc[ni][r] / lrow[wid][row]);
    }
  }
}

// ---------------------------------------------------------------------------
// fp32 -> bf16 weight conversion with padded leading dimension
// ---------------------------------------------------------------------------
__global__ __launch_bounds__(256) void cvt_pad_kernel(
    const float* __restrict__ src, __bf16* __restrict__ dst,
    int rows, int N, int Np)
{
  int i = blockIdx.x * 256 + threadIdx.x;
  if (i >= rows * Np) return;
  int r = i / Np, c = i - r * Np;
  dst[i] = (c < N) ? (__bf16)src[(size_t)r * N + c] : (__bf16)0.0f;
}

// ---------------------------------------------------------------------------
// Embedding gather: fp32 out (residual path) + bf16 mirror (GEMM A path)
// ---------------------------------------------------------------------------
__global__ __launch_bounds__(256) void embed_kernel(
    const int* __restrict__ tok, const float* __restrict__ emb,
    const float* __restrict__ pe, float* __restrict__ outf,
    __bf16* __restrict__ outb, int BL, int E, int L, float scale)
{
  int i = blockIdx.x * 256 + threadIdx.x;
  if (i >= BL * E) return;
  int row = i / E, e = i - row * E;
  int l = row % L;
  float v = emb[(size_t)tok[row] * E + e] * scale + pe[(size_t)l * E + e];
  outf[i] = v;
  outb[i] = (__bf16)v;
}

// ---------------------------------------------------------------------------
// Row LayerNorm, fp32 out + bf16 mirror
// ---------------------------------------------------------------------------
__global__ __launch_bounds__(256) void ln_kernel(
    const float* __restrict__ x, const float* __restrict__ g,
    const float* __restrict__ b, float* __restrict__ outf,
    __bf16* __restrict__ outb, int E)
{
  __shared__ float red[256];
  const int tid = threadIdx.x;
  const float* xr = x + (size_t)blockIdx.x * E;

  float s = 0.0f;
  for (int e = tid; e < E; e += 256) s += xr[e];
  red[tid] = s; __syncthreads();
  for (int off = 128; off > 0; off >>= 1) {
    if (tid < off) red[tid] += red[tid + off];
    __syncthreads();
  }
  float mean = red[0] / (float)E;
  __syncthreads();

  float v = 0.0f;
  for (int e = tid; e < E; e += 256) { float d = xr[e] - mean; v += d * d; }
  red[tid] = v; __syncthreads();
  for (int off = 128; off > 0; off >>= 1) {
    if (tid < off) red[tid] += red[tid + off];
    __syncthreads();
  }
  float inv = rsqrtf(red[0] / (float)E + 1e-5f);

  for (int e = tid; e < E; e += 256) {
    float o = (xr[e] - mean) * inv * g[e] + b[e];
    outf[(size_t)blockIdx.x * E + e] = o;
    outb[(size_t)blockIdx.x * E + e] = (__bf16)o;
  }
}

// ---------------------------------------------------------------------------
// Host orchestration
// ---------------------------------------------------------------------------
static inline void run_gemm(const __bf16* A, const __bf16* Wt, const float* bias,
                            const float* res, float* Cf, __bf16* Cb,
                            int M, int N, int K, int ldw, bool relu, hipStream_t s)
{
  dim3 grid((N + BN - 1) / BN, (M + BM - 1) / BM), block(256);
  if (relu)
    gemm_bf16_wmma<true ><<<grid, block, 0, s>>>(A, Wt, bias, res, Cf, Cb, M, N, K, ldw);
  else
    gemm_bf16_wmma<false><<<grid, block, 0, s>>>(A, Wt, bias, res, Cf, Cb, M, N, K, ldw);
}

static inline void run_cvt(const float* src, __bf16* dst, int rows, int N, int Np,
                           hipStream_t s)
{
  int tot = rows * Np;
  cvt_pad_kernel<<<dim3((tot + 255) / 256), dim3(256), 0, s>>>(src, dst, rows, N, Np);
}

extern "C" void kernel_launch(void* const* d_in, const int* in_sizes, int n_in,
                              void* d_out, int out_size, void* d_ws, size_t ws_size,
                              hipStream_t stream)
{
  (void)in_sizes; (void)n_in; (void)out_size; (void)ws_size;
  const int Bc = 4, Lc = 512, Ec = 512, Hc = 8, DKc = 64, FFc = 2048;
  const int Vc = 30522, Vp = 30528;
  const int BL = Bc * Lc;
  const float emb_scale = sqrtf((float)Ec);
  const float att_scale = 1.0f / sqrtf((float)DKc);

  const int*   src      = (const int*)  d_in[0];
  const int*   trg      = (const int*)  d_in[1];
  const float* src_mask = (const float*)d_in[2];
  const float* trg_mask = (const float*)d_in[3];
  const float* enc_emb  = (const float*)d_in[4];
  const float* dec_emb  = (const float*)d_in[5];
  const float* pe       = (const float*)d_in[6];
  const float* P[43];
  for (int i = 7; i < 43; ++i) P[i] = (const float*)d_in[i];

  char* cur = (char*)d_ws;
  auto allocf = [&](size_t n) { float*  p = (float*) cur; cur += n * sizeof(float);  return p; };
  auto alloch = [&](size_t n) { __bf16* p = (__bf16*)cur; cur += n * sizeof(__bf16); return p; };
  const size_t SZ = (size_t)BL * Ec;

  float* xe = allocf(SZ); float* t0 = allocf(SZ); float* x2 = allocf(SZ);
  float* y0 = allocf(SZ); float* y1 = allocf(SZ); float* y2 = allocf(SZ);
  __bf16* xe_b = alloch(SZ); __bf16* x2_b = alloch(SZ); __bf16* y0_b = alloch(SZ);
  __bf16* y1_b = alloch(SZ); __bf16* y2_b = alloch(SZ); __bf16* x3_b = alloch(SZ);
  __bf16* y3_b = alloch(SZ); __bf16* q_b  = alloch(SZ); __bf16* k_b  = alloch(SZ);
  __bf16* v_b  = alloch(SZ); __bf16* ao_b = alloch(SZ);
  __bf16* ffh_b = alloch((size_t)BL * FFc);
  __bf16* wb[43];
  const int sq[12] = {7, 9, 11, 13, 15, 17, 19, 21, 23, 25, 27, 29};
  for (int i = 0; i < 12; ++i) wb[sq[i]] = alloch((size_t)Ec * Ec);
  wb[33] = alloch((size_t)Ec * FFc); wb[35] = alloch((size_t)FFc * Ec);
  wb[37] = alloch((size_t)Ec * FFc); wb[39] = alloch((size_t)FFc * Ec);
  wb[41] = alloch((size_t)Ec * Vp);

  float* out = (float*)d_out;

  for (int i = 0; i < 12; ++i) run_cvt(P[sq[i]], wb[sq[i]], Ec, Ec, Ec, stream);
  run_cvt(P[33], wb[33], Ec, FFc, FFc, stream);
  run_cvt(P[35], wb[35], FFc, Ec, Ec, stream);
  run_cvt(P[37], wb[37], Ec, FFc, FFc, stream);
  run_cvt(P[39], wb[39], FFc, Ec, Ec, stream);
  run_cvt(P[41], wb[41], Ec, Vc, Vp, stream);

  dim3 eb((BL * Ec + 255) / 256), tb(256);
  dim3 agrid(Lc / 128, Hc, Bc);

  // ---------------- encoder ----------------
  embed_kernel<<<eb, tb, 0, stream>>>(src, enc_emb, pe, xe, xe_b, BL, Ec, Lc, emb_scale);
  run_gemm(xe_b, wb[7],  P[8],  nullptr, nullptr, q_b, BL, Ec, Ec, Ec, false, stream);
  run_gemm(xe_b, wb[9],  P[10], nullptr, nullptr, k_b, BL, Ec, Ec, Ec, false, stream);
  run_gemm(xe_b, wb[11], P[12], nullptr, nullptr, v_b, BL, Ec, Ec, Ec, false, stream);
  flash_attn_wmma<<<agrid, tb, 0, stream>>>(q_b, k_b, v_b, src_mask, ao_b, Lc, Ec, DKc, att_scale);
  run_gemm(ao_b, wb[13], P[14], xe, t0, nullptr, BL, Ec, Ec, Ec, false, stream);
  ln_kernel<<<dim3(BL), tb, 0, stream>>>(t0, P[31], P[32], x2, x2_b, Ec);
  run_gemm(x2_b,  wb[33], P[34], nullptr, nullptr, ffh_b, BL, FFc, Ec, FFc, true,  stream);
  run_gemm(ffh_b, wb[35], P[36], x2,      nullptr, x3_b,  BL, Ec, FFc, Ec, false, stream);

  // ---------------- decoder ----------------
  embed_kernel<<<eb, tb, 0, stream>>>(trg, dec_emb, pe, y0, y0_b, BL, Ec, Lc, emb_scale);
  run_gemm(y0_b, wb[15], P[16], nullptr, nullptr, q_b, BL, Ec, Ec, Ec, false, stream);
  run_gemm(y0_b, wb[17], P[18], nullptr, nullptr, k_b, BL, Ec, Ec, Ec, false, stream);
  run_gemm(y0_b, wb[19], P[20], nullptr, nullptr, v_b, BL, Ec, Ec, Ec, false, stream);
  flash_attn_wmma<<<agrid, tb, 0, stream>>>(q_b, k_b, v_b, trg_mask, ao_b, Lc, Ec, DKc, att_scale);
  run_gemm(ao_b, wb[21], P[22], y0, y1, y1_b, BL, Ec, Ec, Ec, false, stream);
  run_gemm(y1_b, wb[23], P[24], nullptr, nullptr, q_b, BL, Ec, Ec, Ec, false, stream);
  run_gemm(x3_b, wb[25], P[26], nullptr, nullptr, k_b, BL, Ec, Ec, Ec, false, stream);
  run_gemm(x3_b, wb[27], P[28], nullptr, nullptr, v_b, BL, Ec, Ec, Ec, false, stream);
  flash_attn_wmma<<<agrid, tb, 0, stream>>>(q_b, k_b, v_b, src_mask, ao_b, Lc, Ec, DKc, att_scale);
  run_gemm(ao_b, wb[29], P[30], y1, y2, y2_b, BL, Ec, Ec, Ec, false, stream);
  run_gemm(y2_b,  wb[37], P[38], nullptr, nullptr, ffh_b, BL, FFc, Ec, FFc, true,  stream);
  run_gemm(ffh_b, wb[39], P[40], y2,      nullptr, y3_b,  BL, Ec, FFc, Ec, false, stream);

  // ---------------- vocab projection (fp32 to d_out) ----------------
  run_gemm(y3_b, wb[41], P[42], nullptr, out, nullptr, BL, Vc, Ec, Vp, false, stream);
}
